// GIN_33578054320560
// MI455X (gfx1250) — matrix operand order repaired and checked by reading, not compiled, not run on hardware
//
#include <hip/hip_runtime.h>
#include <hip/hip_bf16.h>
#include <stdint.h>

// ---------------- problem constants (match reference) ----------------
#define NNODES  100000
#define NEDGES  1600000
#define FIN     64
#define FF      128
#define NCONV   4
#define NGRAPHS 2000
#define CHID    256
#define CAT     (FF * (NCONV + 2))   // 768
#define BN_EPS  1e-5f
#define SLOPE   0.01f

typedef __attribute__((ext_vector_type(16))) _Float16 v16h;
typedef __attribute__((ext_vector_type(8)))  _Float16 v8h;
typedef __attribute__((ext_vector_type(4)))  _Float16 v4h;
typedef __attribute__((ext_vector_type(2)))  _Float16 v2h;
typedef __attribute__((ext_vector_type(8)))  float    v8f;

union HF16 { v16h v; v8h h[2]; };

// ---------------------------------------------------------------------
// zero-fill (float4 wide; n must be a multiple of 4 — all our buffers are)
__global__ void zero_kernel(float4* __restrict__ p, long n4) {
    long t = (long)blockIdx.x * blockDim.x + threadIdx.x;
    if (t < n4) p[t] = make_float4(0.f, 0.f, 0.f, 0.f);
}

// weight prep: W[K,NN] f32 row-major  ->  Wt[NN,K] f16 row-major (transposed)
__global__ void wtrans_kernel(const float* __restrict__ W, _Float16* __restrict__ Wt,
                              int K, int NN, long total) {
    long t = (long)blockIdx.x * blockDim.x + threadIdx.x;
    if (t >= total) return;
    long k = t / NN;
    int  c = (int)(t - k * NN);
    Wt[(size_t)c * K + k] = (_Float16)W[t];
}

// edge aggregation, f32 source (layer 1: x). 4 features per thread (16B load).
__global__ void agg_f32_kernel(const float* __restrict__ h, int ld,
                               const int* __restrict__ srcv, const int* __restrict__ dstv,
                               float* __restrict__ agg, int F, int shift, long total) {
    long t = (long)blockIdx.x * blockDim.x + threadIdx.x;
    if (t >= total) return;
    long e = t >> shift;
    int  c = (int)(t & ((1 << shift) - 1)) << 2;
    int  s = srcv[e], d = dstv[e];
    float4 val = *(const float4*)(h + (size_t)s * ld + c);
    float* p = agg + (size_t)d * F + c;
    __hip_atomic_fetch_add(p + 0, val.x, __ATOMIC_RELAXED, __HIP_MEMORY_SCOPE_AGENT);
    __hip_atomic_fetch_add(p + 1, val.y, __ATOMIC_RELAXED, __HIP_MEMORY_SCOPE_AGENT);
    __hip_atomic_fetch_add(p + 2, val.z, __ATOMIC_RELAXED, __HIP_MEMORY_SCOPE_AGENT);
    __hip_atomic_fetch_add(p + 3, val.w, __ATOMIC_RELAXED, __HIP_MEMORY_SCOPE_AGENT);
}

// edge aggregation, f16 source (concat slice). 8 features per thread (b128 load).
__global__ void agg_f16_kernel(const _Float16* __restrict__ h, int ld, int col_off,
                               const int* __restrict__ srcv, const int* __restrict__ dstv,
                               float* __restrict__ agg, int F, int shift, long total) {
    long t = (long)blockIdx.x * blockDim.x + threadIdx.x;
    if (t >= total) return;
    long e = t >> shift;
    int  c = (int)(t & ((1 << shift) - 1)) << 3;
    int  s = srcv[e], d = dstv[e];
    v8h val = *(const v8h*)(h + (size_t)s * ld + col_off + c);
    float* p = agg + (size_t)d * F + c;
#pragma unroll
    for (int j = 0; j < 8; j++)
        __hip_atomic_fetch_add(p + j, (float)val[j],
                               __ATOMIC_RELAXED, __HIP_MEMORY_SCOPE_AGENT);
}

// z = (1+eps)*h + agg  -> f16 GEMM input buffer [N, F]; 4 features per thread
__global__ void prep_kernel(const float* __restrict__ xf, const _Float16* __restrict__ hf,
                            int ld_in, int col_off, const float* __restrict__ agg,
                            const float* __restrict__ eps_arr, int eps_idx,
                            _Float16* __restrict__ out16, int logF4, long total4) {
    long t = (long)blockIdx.x * blockDim.x + threadIdx.x;
    if (t >= total4) return;
    long node = t >> logF4;
    int  c = (int)(t & ((1 << logF4) - 1)) << 2;
    float e = 1.0f + eps_arr[eps_idx];
    float4 a = *(const float4*)(agg + (t << 2));
    float h0, h1, h2, h3;
    if (xf) {
        float4 hv = *(const float4*)(xf + node * (long)ld_in + c);
        h0 = hv.x; h1 = hv.y; h2 = hv.z; h3 = hv.w;
    } else {
        v4h hv = *(const v4h*)(hf + node * (long)ld_in + col_off + c);
        h0 = (float)hv[0]; h1 = (float)hv[1]; h2 = (float)hv[2]; h3 = (float)hv[3];
    }
    v4h o;
    o[0] = (_Float16)(e * h0 + a.x);
    o[1] = (_Float16)(e * h1 + a.y);
    o[2] = (_Float16)(e * h2 + a.z);
    o[3] = (_Float16)(e * h3 + a.w);
    *(v4h*)(out16 + (t << 2)) = o;
}

// ---------------------------------------------------------------------
// WMMA GEMM: Out[M,Fout](+col_off) = epilogue( A[M,K]f16 @ Bt[Fout,K]^T f16 )
// one wave per 16x16 tile; grid = (M/32, Fout/16); block = 64 (2 waves)
// mode 0: +bias, BN(eval), leaky(leaky(.))   (GIN layer)
// mode 1: +bias                              (classify_1)
// mode 2: +bias, leaky                       (classifier hidden)
__global__ __launch_bounds__(64) void wmma_gemm_kernel(
    const _Float16* __restrict__ A, int lda,
    const _Float16* __restrict__ Bt, int K,
    const float* __restrict__ bias,
    const float* __restrict__ bn_g, const float* __restrict__ bn_be,
    const float* __restrict__ bn_m, const float* __restrict__ bn_v,
    _Float16* __restrict__ Out, int ldo, int col_off, int mode) {
    const int lane    = threadIdx.x & 31;
    const int wave    = threadIdx.x >> 5;
    const int rowBase = (blockIdx.x * 2 + wave) * 16;
    const int colBase = blockIdx.y * 16;
    const int r       = lane & 15;
    const int hi      = lane >> 4;

    const _Float16* arow = A  + (size_t)(rowBase + r) * lda;
    const _Float16* bcol = Bt + (size_t)(colBase + r) * K;

    v8f acc = {};
    for (int kb = 0; kb < K; kb += 32) {
        // A frag (16x32 f16): lane holds row r; halves 0-7 = K kb+hi*8..+7,
        // halves 8-15 = K kb+16+hi*8..+7  (per ISA 16-bit A layout)
        HF16 af;
        af.h[0] = *(const v8h*)(arow + kb + hi * 8);
        af.h[1] = *(const v8h*)(arow + kb + 16 + hi * 8);
        // B frag (32x16 f16): lane holds col r; same K order as A frag
        HF16 bfu;
        bfu.h[0] = *(const v8h*)(bcol + kb + hi * 8);
        bfu.h[1] = *(const v8h*)(bcol + kb + 16 + hi * 8);
        v16h bf = bfu.v;
        acc = __builtin_amdgcn_wmma_f32_16x16x32_f16(
            false, af.v, false, bf, (short)0, acc, false, false);
    }

    const int c  = colBase + r;
    const float bb = bias[c];
    float scale = 1.0f, mm = 0.0f, bbe = 0.0f;
    if (mode == 0) {
        scale = bn_g[c] * rsqrtf(bn_v[c] + BN_EPS);
        mm    = bn_m[c];
        bbe   = bn_be[c];
    }
    _Float16* ocol = Out + col_off + c;
#pragma unroll
    for (int j = 0; j < 8; j++) {
        float v = acc[j] + bb;
        if (mode == 0) {
            v = (v - mm) * scale + bbe;
            v = v > 0.0f ? v : (SLOPE * SLOPE) * v;   // inner + outer leaky
        } else if (mode == 2) {
            v = v > 0.0f ? v : SLOPE * v;
        }
        int m = rowBase + hi * 8 + j;                  // D layout: vgpr j -> row j + hi*8
        ocol[(size_t)m * ldo] = (_Float16)v;
    }
}

// ---------------------------------------------------------------------
// global_add_pool: hpool[batch[n], c] += h5[n, c]; 4 features per thread
__global__ void pool_kernel(const _Float16* __restrict__ concat16, const int* __restrict__ batch,
                            float* __restrict__ hpool, long total4) {
    long t = (long)blockIdx.x * blockDim.x + threadIdx.x;
    if (t >= total4) return;
    long node = t >> 5;                       // 32 groups of 4 per node
    int  c = (int)(t & 31) << 2;
    v4h hv = *(const v4h*)(concat16 + node * CAT + 4 * FF + c);
    float* p = hpool + (size_t)batch[node] * FF + c;
#pragma unroll
    for (int j = 0; j < 4; j++)
        __hip_atomic_fetch_add(p + j, (float)hv[j],
                               __ATOMIC_RELAXED, __HIP_MEMORY_SCOPE_AGENT);
}

// broadcast pooled features back per node into concat slice 5; 4/thread
__global__ void bcast_kernel(_Float16* __restrict__ concat16, const int* __restrict__ batch,
                             const float* __restrict__ hpool, long total4) {
    long t = (long)blockIdx.x * blockDim.x + threadIdx.x;
    if (t >= total4) return;
    long node = t >> 5;
    int  c = (int)(t & 31) << 2;
    float4 pv = *(const float4*)(hpool + (size_t)batch[node] * FF + c);
    v4h o;
    o[0] = (_Float16)pv.x; o[1] = (_Float16)pv.y;
    o[2] = (_Float16)pv.z; o[3] = (_Float16)pv.w;
    *(v4h*)(concat16 + node * CAT + 5 * FF + c) = o;
}

// final: out[n] = sigmoid( act[n,:] . w_f + b_f )
__global__ void final_kernel(const _Float16* __restrict__ act, const float* __restrict__ wf,
                             const float* __restrict__ bf, float* __restrict__ out, int n) {
    int i = blockIdx.x * blockDim.x + threadIdx.x;
    if (i >= n) return;
    const _Float16* a = act + (size_t)i * CHID;
    float s = bf[0];
#pragma unroll 8
    for (int j = 0; j < CHID; j++) s += (float)a[j] * wf[j];
    out[i] = 1.0f / (1.0f + __expf(-s));
}

// ---------------------------------------------------------------------
extern "C" void kernel_launch(void* const* d_in, const int* in_sizes, int n_in,
                              void* d_out, int out_size, void* d_ws, size_t ws_size,
                              hipStream_t stream) {
    const float* x     = (const float*)d_in[0];
    const int*   eidx  = (const int*)  d_in[1];
    const int*   batch = (const int*)  d_in[2];
    const float* w1    = (const float*)d_in[3];
    const float* b1    = (const float*)d_in[4];
    const float* g1    = (const float*)d_in[5];
    const float* be1   = (const float*)d_in[6];
    const float* m1    = (const float*)d_in[7];
    const float* v1    = (const float*)d_in[8];
    const float* eps1  = (const float*)d_in[9];
    const float* cw    = (const float*)d_in[10];
    const float* cb    = (const float*)d_in[11];
    const float* cg    = (const float*)d_in[12];
    const float* cbe   = (const float*)d_in[13];
    const float* cm    = (const float*)d_in[14];
    const float* cv    = (const float*)d_in[15];
    const float* ceps  = (const float*)d_in[16];
    const float* w_c1  = (const float*)d_in[17];
    const float* b_c1  = (const float*)d_in[18];
    const float* w_cls = (const float*)d_in[19];
    const float* b_cls = (const float*)d_in[20];
    const float* w_f   = (const float*)d_in[21];
    const float* b_f   = (const float*)d_in[22];
    (void)in_sizes; (void)n_in; (void)out_size; (void)ws_size;

    const int* srcv = eidx;
    const int* dstv = eidx + NEDGES;

    // -------- workspace carve (deterministic) --------
    uint8_t* base = (uint8_t*)d_ws;
    size_t off = 0;
    auto carve = [&](size_t bytes) -> void* {
        off = (off + 255) & ~(size_t)255;
        void* p = base + off;
        off += bytes;
        return p;
    };
    _Float16* concat16 = (_Float16*)carve((size_t)NNODES * CAT * 2);   // [N,768] f16
    float*    agg      = (float*)   carve((size_t)NNODES * FF * 4);    // [N,128] f32
    _Float16* xin16    = (_Float16*)carve((size_t)NNODES * FF * 2);    // GEMM input
    _Float16* actA     = (_Float16*)carve((size_t)NNODES * CHID * 2);
    _Float16* actB     = (_Float16*)carve((size_t)NNODES * CHID * 2);
    float*    hpool    = (float*)   carve((size_t)NGRAPHS * FF * 4);
    _Float16* w1t      = (_Float16*)carve((size_t)FIN * FF * 2);
    _Float16* cwt      = (_Float16*)carve((size_t)NCONV * FF * FF * 2);
    _Float16* wc1t     = (_Float16*)carve((size_t)CAT * CHID * 2);
    _Float16* wclst    = (_Float16*)carve((size_t)2 * CHID * CHID * 2);

    auto nb = [](long n) { return (unsigned)((n + 255) / 256); };

    // -------- weight prep (f32 -> transposed f16) --------
    wtrans_kernel<<<nb((long)FIN * FF), 256, 0, stream>>>(w1, w1t, FIN, FF, (long)FIN * FF);
    for (int i = 0; i < NCONV; i++)
        wtrans_kernel<<<nb((long)FF * FF), 256, 0, stream>>>(
            cw + (size_t)i * FF * FF, cwt + (size_t)i * FF * FF, FF, FF, (long)FF * FF);
    wtrans_kernel<<<nb((long)CAT * CHID), 256, 0, stream>>>(w_c1, wc1t, CAT, CHID, (long)CAT * CHID);
    for (int i = 0; i < 2; i++)
        wtrans_kernel<<<nb((long)CHID * CHID), 256, 0, stream>>>(
            w_cls + (size_t)i * CHID * CHID, wclst + (size_t)i * CHID * CHID,
            CHID, CHID, (long)CHID * CHID);

    // -------- layer 1: agg over x (F=64), GEMM 64->128 --------
    {
        long na4 = (long)NNODES * FIN / 4;
        zero_kernel<<<nb(na4), 256, 0, stream>>>((float4*)agg, na4);
        long tw = (long)NEDGES * (FIN / 4);
        agg_f32_kernel<<<nb(tw), 256, 0, stream>>>(x, FIN, srcv, dstv, agg, FIN, 4, tw);
        long tp = (long)NNODES * FIN / 4;
        prep_kernel<<<nb(tp), 256, 0, stream>>>(x, (const _Float16*)nullptr, FIN, 0,
                                                agg, eps1, 0, xin16, 4, tp);
        dim3 g(NNODES / 32, FF / 16);
        wmma_gemm_kernel<<<g, 64, 0, stream>>>(xin16, FIN, w1t, FIN,
                                               b1, g1, be1, m1, v1,
                                               concat16, CAT, 0, 0);
    }

    // -------- conv layers 2..5: agg over concat slice i, GEMM 128->128 --------
    for (int i = 0; i < NCONV; i++) {
        long na4 = (long)NNODES * FF / 4;
        zero_kernel<<<nb(na4), 256, 0, stream>>>((float4*)agg, na4);
        long tw = (long)NEDGES * (FF / 8);          // 16 lanes/edge, b128 gathers
        agg_f16_kernel<<<nb(tw), 256, 0, stream>>>(concat16, CAT, i * FF,
                                                   srcv, dstv, agg, FF, 4, tw);
        long tp = (long)NNODES * FF / 4;
        prep_kernel<<<nb(tp), 256, 0, stream>>>((const float*)nullptr, concat16, CAT, i * FF,
                                                agg, ceps, i, xin16, 5, tp);
        dim3 g(NNODES / 32, FF / 16);
        wmma_gemm_kernel<<<g, 64, 0, stream>>>(xin16, FF, cwt + (size_t)i * FF * FF, FF,
                                               cb + (size_t)i * FF, cg + (size_t)i * FF,
                                               cbe + (size_t)i * FF, cm + (size_t)i * FF,
                                               cv + (size_t)i * FF,
                                               concat16, CAT, (i + 1) * FF, 0);
    }

    // -------- global_add_pool + broadcast into concat slice 5 --------
    {
        long np4 = (long)NGRAPHS * FF / 4;
        zero_kernel<<<nb(np4), 256, 0, stream>>>((float4*)hpool, np4);
        long tp = (long)NNODES * FF / 4;
        pool_kernel<<<nb(tp), 256, 0, stream>>>(concat16, batch, hpool, tp);
        bcast_kernel<<<nb(tp), 256, 0, stream>>>(concat16, batch, hpool, tp);
    }

    // -------- classifier --------
    {
        dim3 g(NNODES / 32, CHID / 16);
        // h = concat @ w_c1 + b_c1   (no activation)
        wmma_gemm_kernel<<<g, 64, 0, stream>>>(concat16, CAT, wc1t, CAT,
                                               b_c1, nullptr, nullptr, nullptr, nullptr,
                                               actA, CHID, 0, 1);
        // h = leaky(h @ w_cls0 + b0)
        wmma_gemm_kernel<<<g, 64, 0, stream>>>(actA, CHID, wclst, CHID,
                                               b_cls, nullptr, nullptr, nullptr, nullptr,
                                               actB, CHID, 0, 2);
        // h = leaky(h @ w_cls1 + b1)
        wmma_gemm_kernel<<<g, 64, 0, stream>>>(actB, CHID, wclst + (size_t)CHID * CHID, CHID,
                                               b_cls + CHID, nullptr, nullptr, nullptr, nullptr,
                                               actA, CHID, 0, 2);
        // out = sigmoid(h @ w_f + b_f)
        final_kernel<<<nb(NNODES), 256, 0, stream>>>(actA, w_f, b_f, (float*)d_out, NNODES);
    }
}
